// SpiralNet_11819749998924
// MI455X (gfx1250) — matrix operand order repaired and checked by reading, not compiled, not run on hardware
//
#include <hip/hip_runtime.h>
#include <hip/hip_bf16.h>
#include <stdint.h>

#define SEQ 12

typedef __attribute__((ext_vector_type(16))) __bf16 v16bf;
typedef __attribute__((ext_vector_type(8)))  float  v8f;

union Frag16 { v16bf v; uint4 q[2]; };

__device__ __forceinline__ unsigned short f32_to_bf16_rne(float f) {
  unsigned int u = __float_as_uint(f);
  unsigned int lsb = (u >> 16) & 1u;
  u += 0x7fffu + lsb;
  return (unsigned short)(u >> 16);
}

// ---------------- preprocessing ----------------

__global__ __launch_bounds__(256)
void cvt_f32_to_bf16(const float* __restrict__ src, unsigned short* __restrict__ dst, int count) {
  int i = blockIdx.x * blockDim.x + threadIdx.x;
  if (i < count) dst[i] = f32_to_bf16_rne(src[i]);
}

// W [K][Cout] fp32 (row-major) -> Wt [Cout][K] bf16 (transposed)
__global__ __launch_bounds__(256)
void cvt_transpose_w(const float* __restrict__ W, unsigned short* __restrict__ Wt, int K, int Cout) {
  int i = blockIdx.x * blockDim.x + threadIdx.x;
  if (i < K * Cout) {
    int k = i / Cout, co = i - k * Cout;
    Wt[co * K + k] = f32_to_bf16_rne(W[i]);
  }
}

// ---------------- fused gather + GEMM + bias + ELU layer ----------------
//
// Hin:  bf16 activation table [n][C_IN]
// sidx: int32 [n][SEQ]
// Wt:   bf16 transposed weights [C_OUT][K],  K = SEQ*C_IN
// out:  bf16 [n][C_OUT] (intermediate) or f32 [n][C_OUT] (final)
//
// 8 waves / block, each wave computes 2 row-tiles (32 rows) x all column tiles.
// Weights staged in LDS, padded stride (K+8) bf16 -> stride = 4 dwords mod 64 banks.
// Spiral indices hoisted to registers; K-loop fully unrolled so all segment
// selects and LDS offsets are compile-time constants.

template<int C_IN, int C_OUT, bool ACT, bool OUT_F32>
__global__ __launch_bounds__(256)
void spiral_layer(const unsigned short* __restrict__ Hin,
                  const int* __restrict__ sidx,
                  const unsigned short* __restrict__ Wt,
                  const float* __restrict__ bias,
                  unsigned short* __restrict__ HoutBf,
                  float* __restrict__ HoutF,
                  int n) {
  constexpr int K  = SEQ * C_IN;   // 384 or 768, divisible by 32
  constexpr int KT = K / 32;
  constexpr int CT = C_OUT / 16;
  constexpr int RT = 2;            // row tiles per wave (32 rows)
  constexpr int KP = K + 8;        // padded LDS row (bf16 elems); (K+8)/8 integral

  extern __shared__ unsigned short sW[];

  // ---- stage Wt -> LDS (128-bit chunks) ----
  {
    constexpr int KW  = K / 8;     // uint4 chunks per (unpadded) row
    constexpr int KWP = KP / 8;    // uint4 chunks per padded row
    const uint4* gw = (const uint4*)Wt;
    uint4* lw = (uint4*)sW;
    for (int c = threadIdx.x; c < C_OUT * KW; c += 256) {
      int co = c / KW, t8 = c - co * KW;
      lw[co * KWP + t8] = gw[c];
    }
  }
  __syncthreads();

  const int lane  = threadIdx.x & 31;
  const int half  = lane >> 4;       // 0: K-chunks {0..7,16..23}; 1: {8..15,24..31}
  const int half8 = half * 8;
  const int l16   = lane & 15;
  const int wave  = threadIdx.x >> 5;
  const int rowbase = (blockIdx.x * 8 + wave) * (RT * 16);
  if (rowbase >= n) return;          // wave-uniform; EXEC stays all-ones below

  v8f acc[RT][CT];
#pragma unroll
  for (int rt = 0; rt < RT; ++rt)
#pragma unroll
    for (int ct = 0; ct < CT; ++ct)
      acc[rt][ct] = {};

  // clamped rows for the A-side gather (stores are guarded later)
  int rowA[RT];
#pragma unroll
  for (int rt = 0; rt < RT; ++rt) {
    int r = rowbase + rt * 16 + l16;
    rowA[rt] = (r < n) ? r : (n - 1);
  }

  // hoist all SEQ spiral indices per row into registers (3 x int4 per row tile)
  int I[RT][SEQ];
#pragma unroll
  for (int rt = 0; rt < RT; ++rt) {
    const int4* ip = (const int4*)(sidx + rowA[rt] * SEQ);   // 48B rows -> 16B aligned
    int4 q0 = ip[0], q1 = ip[1], q2 = ip[2];
    I[rt][0] = q0.x; I[rt][1]  = q0.y; I[rt][2]  = q0.z; I[rt][3]  = q0.w;
    I[rt][4] = q1.x; I[rt][5]  = q1.y; I[rt][6]  = q1.z; I[rt][7]  = q1.w;
    I[rt][8] = q2.x; I[rt][9]  = q2.y; I[rt][10] = q2.z; I[rt][11] = q2.w;
  }

  // per-lane LDS base for B fragments
  const unsigned short* bbase = sW + l16 * KP + half8;

#pragma unroll
  for (int kt = 0; kt < KT; ++kt) {
    // Segment of each 8-wide K-chunk is lane-independent: the half*8 term can
    // never cross a C_IN boundary (chunk starts 8-aligned, C_IN in {32,64}).
    const int seg0 = (kt * 32) / C_IN;            // compile-time after unroll
    const int seg1 = (kt * 32 + 16) / C_IN;       // compile-time after unroll
    const int off0 = kt * 32      - seg0 * C_IN + half8;
    const int off1 = kt * 32 + 16 - seg1 * C_IN + half8;

    Frag16 a[RT];
#pragma unroll
    for (int rt = 0; rt < RT; ++rt) {
      a[rt].q[0] = *(const uint4*)(Hin + I[rt][seg0] * C_IN + off0);  // 8 bf16, 16B aligned
      a[rt].q[1] = *(const uint4*)(Hin + I[rt][seg1] * C_IN + off1);
    }

#pragma unroll
    for (int ct = 0; ct < CT; ++ct) {
      Frag16 b;
      const uint4* p = (const uint4*)(bbase + ct * 16 * KP + kt * 32);
      b.q[0] = p[0];
      b.q[1] = p[2];
#pragma unroll
      for (int rt = 0; rt < RT; ++rt)
        acc[rt][ct] = __builtin_amdgcn_wmma_f32_16x16x32_bf16(
            false, a[rt].v, false, b.v, (short)0, acc[rt][ct], false, false);
    }
  }

  // ---- epilogue: bias + (ELU) + store ----
  // C/D layout: VGPR v, lanes 0-15 -> (M=v, N=lane); lanes 16-31 -> (M=v+8, N=lane-16)
#pragma unroll
  for (int rt = 0; rt < RT; ++rt) {
#pragma unroll
    for (int ct = 0; ct < CT; ++ct) {
      const int col = ct * 16 + l16;
      const float bv = bias[col];
#pragma unroll
      for (int v = 0; v < 8; ++v) {
        const int r = rowbase + rt * 16 + v + half * 8;
        if (r < n) {
          float xv = acc[rt][ct][v] + bv;
          if (ACT) xv = (xv > 0.0f) ? xv : expm1f(xv);   // ELU(alpha=1)
          if (OUT_F32) HoutF[r * C_OUT + col] = xv;
          else         HoutBf[r * C_OUT + col] = f32_to_bf16_rne(xv);
        }
      }
    }
  }
}

// ---------------- host launcher ----------------

extern "C" void kernel_launch(void* const* d_in, const int* in_sizes, int n_in,
                              void* d_out, int out_size, void* d_ws, size_t ws_size,
                              hipStream_t stream) {
  const float* x   = (const float*)d_in[0];   // [n,32,1]
  const int*   idx = (const int*)  d_in[1];   // [n,12]
  const float* W0  = (const float*)d_in[2];   // [384,64]
  const float* b0  = (const float*)d_in[3];
  const float* W1  = (const float*)d_in[4];   // [768,64]
  const float* b1  = (const float*)d_in[5];
  const float* W2  = (const float*)d_in[6];   // [768,32]
  const float* b2  = (const float*)d_in[7];
  float* out = (float*)d_out;                 // [n,32,1] fp32
  const int n = in_sizes[0] / 32;

  // workspace carve (256B aligned)
  char* ws = (char*)d_ws;
  size_t o = 0;
  auto carve = [&](size_t bytes) { char* p = ws + o; o += (bytes + 255) & ~(size_t)255; return p; };
  unsigned short* H0  = (unsigned short*)carve((size_t)n * 32 * 2);
  unsigned short* H1  = (unsigned short*)carve((size_t)n * 64 * 2);
  unsigned short* H2  = (unsigned short*)carve((size_t)n * 64 * 2);
  unsigned short* Wt0 = (unsigned short*)carve((size_t)384 * 64 * 2);
  unsigned short* Wt1 = (unsigned short*)carve((size_t)768 * 64 * 2);
  unsigned short* Wt2 = (unsigned short*)carve((size_t)768 * 32 * 2);

  // preprocess: fp32 -> bf16, weights transposed to [Cout][K]
  cvt_f32_to_bf16<<<(n * 32 + 255) / 256, 256, 0, stream>>>(x, H0, n * 32);
  cvt_transpose_w<<<(384 * 64 + 255) / 256, 256, 0, stream>>>(W0, Wt0, 384, 64);
  cvt_transpose_w<<<(768 * 64 + 255) / 256, 256, 0, stream>>>(W1, Wt1, 768, 64);
  cvt_transpose_w<<<(768 * 32 + 255) / 256, 256, 0, stream>>>(W2, Wt2, 768, 32);

  const int grid = (n + 255) / 256;                 // 256 rows per block
  const size_t sh0 = (size_t)64 * (384 + 8) * 2;    // 50176 B
  const size_t sh1 = (size_t)64 * (768 + 8) * 2;    // 99328 B
  const size_t sh2 = (size_t)32 * (768 + 8) * 2;    // 49664 B

  (void)hipFuncSetAttribute((const void*)spiral_layer<32, 64, true,  false>,
                            hipFuncAttributeMaxDynamicSharedMemorySize, (int)sh0);
  (void)hipFuncSetAttribute((const void*)spiral_layer<64, 64, true,  false>,
                            hipFuncAttributeMaxDynamicSharedMemorySize, (int)sh1);
  (void)hipFuncSetAttribute((const void*)spiral_layer<64, 32, false, true>,
                            hipFuncAttributeMaxDynamicSharedMemorySize, (int)sh2);

  spiral_layer<32, 64, true,  false><<<grid, 256, sh0, stream>>>(H0, idx, Wt0, b0, H1, nullptr, n);
  spiral_layer<64, 64, true,  false><<<grid, 256, sh1, stream>>>(H1, idx, Wt1, b1, H2, nullptr, n);
  spiral_layer<64, 32, false, true ><<<grid, 256, sh2, stream>>>(H2, idx, Wt2, b2, nullptr, out, n);
}
